// spiking_coESN_rescaled_66915590472000
// MI455X (gfx1250) — compile-verified, matrix operand order checked by reading, not run on hardware
//
#include <hip/hip_runtime.h>

typedef __attribute__((ext_vector_type(16))) __bf16       v16bf;
typedef __attribute__((ext_vector_type(8)))  float        v8f;
typedef __attribute__((ext_vector_type(4)))  unsigned int v4u;

#define NHID     1024
#define SEQL     1024
#define BATCH    2048
#define MT       16      // batch rows per block (one WMMA M tile)
#define NTW      4       // N-tiles (of 16 cols) per wave
#define NTHREADS 512     // 16 waves cover all 64 N-tiles
#define NKT      32      // number of K-tiles (32 wide each)
#define PST      1026    // float2 state row stride (elements): halves -> disjoint banks
#define LST      1028    // f32 state row stride: halves -> disjoint banks

union ABFrag { v16bf bf; v4u u4[2]; };
struct __align__(8) F2 { float x, y; };   // trivially-constructible float2 for __shared__

__device__ __forceinline__ unsigned short f2bf(float f) {
  unsigned u = __builtin_bit_cast(unsigned, f);
  u += 0x7FFFu + ((u >> 16) & 1u);          // round-to-nearest-even
  return (unsigned short)(u >> 16);
}

// XOR swizzle (8-element chunks) so 16 rows at equal K offset hit 16 distinct
// 4-bank groups for the ds_load_b128 A-fragment reads.
__device__ __forceinline__ int sw_addr(int row, int k) {
  return row * NHID + ((((k >> 3) ^ row) << 3) | (k & 7));
}

// One-time repack: h2h (f32 KxN row-major) -> bf16 WMMA B-fragments, 32
// contiguous bytes per lane per (nt,kt) so GEMM loads are global_load_b128.
__global__ void coesn_prep_b(const float* __restrict__ h2h,
                             unsigned short* __restrict__ bpack) {
  int idx  = blockIdx.x * 256 + threadIdx.x;     // 0 .. NHID*NHID-1
  int j    = idx & 15;
  int lane = (idx >> 4) & 31;
  int kt   = (idx >> 9) & 31;
  int nt   = idx >> 14;                          // 0..63
  int n    = nt * 16 + (lane & 15);
  int hf   = lane >> 4;
  int k    = kt * 32 + ((j < 8) ? (hf * 8 + j) : (16 + hf * 8 + (j - 8)));
  bpack[idx] = f2bf(h2h[k * NHID + n]);
}

__device__ __forceinline__ void load_a(const unsigned short* sc, int lrow, int hf,
                                       int kt, ABFrag& a) {
  const unsigned short* ap0 = sc + sw_addr(lrow, kt * 32 + hf * 8);
  const unsigned short* ap1 = sc + sw_addr(lrow, kt * 32 + 16 + hf * 8);
  a.u4[0] = *(const v4u*)ap0;
  a.u4[1] = *(const v4u*)ap1;
}

__device__ __forceinline__ void load_b(const unsigned short* __restrict__ bpack,
                                       int wave, int lane, int kt, ABFrag b[NTW]) {
#pragma unroll
  for (int j = 0; j < NTW; ++j) {
    int nt = wave * NTW + j;
    const v4u* bp = (const v4u*)bpack + (size_t)((nt * NKT + kt) * 32 + lane) * 2;
    b[j].u4[0] = bp[0];
    b[j].u4[1] = bp[1];
  }
}

__device__ __forceinline__ void do_wmma(const ABFrag& a, const ABFrag b[NTW],
                                        v8f acc[NTW]) {
#pragma unroll
  for (int j = 0; j < NTW; ++j)
    acc[j] = __builtin_amdgcn_wmma_f32_16x16x32_bf16(
        false, a.bf, false, b[j].bf, (short)0, acc[j], false, false);
}

__global__ __launch_bounds__(NTHREADS)
void coesn_persistent(const float* __restrict__ x,      // (B, L, 1)
                      const float* __restrict__ x2h,    // (1, NHID)
                      const float* __restrict__ bias,
                      const float* __restrict__ gamma,
                      const float* __restrict__ eps,
                      const float* __restrict__ gainp,  // scalar
                      const unsigned short* __restrict__ bpack,
                      float* __restrict__ out) {        // (B, 2*NHID)
  // ~225 KB of the 320 KB WGP LDS: spikes + three offloaded state arrays.
  __shared__ __align__(16) unsigned short sbuf[MT * NHID]; // bf16 spikes, 32 KB
  __shared__ F2    rs_l[MT * PST];                         // {rf, fts}, 128 KB
  __shared__ float lv_l[MT * LST];                         // lif_v,      64 KB

  const int tid  = threadIdx.x;
  const int wave = tid >> 5;
  const int lane = tid & 31;
  const int hf   = lane >> 4;       // 0: lanes 0-15, 1: lanes 16-31
  const int lrow = lane & 15;
  const int rb   = blockIdx.x * MT; // first batch row owned by this block

  for (int i = tid; i < MT * NHID; i += NTHREADS) sbuf[i] = 0;
  for (int i = tid; i < MT * PST;  i += NTHREADS) rs_l[i] = F2{0.f, 0.f};
  for (int i = tid; i < MT * LST;  i += NTHREADS) lv_l[i] = 0.f;

  // Per-lane column constants for each owned N-tile (column n = tile*16 + lrow).
  float gam[NTW], ev[NTW], bv[NTW], xw[NTW];
#pragma unroll
  for (int j = 0; j < NTW; ++j) {
    int n = (wave * NTW + j) * 16 + lrow;
    gam[j] = gamma[n]; ev[j] = eps[n]; bv[j] = bias[n]; xw[j] = x2h[n];
  }
  const float sg = gainp[0];

  // Hot per-element state kept in VGPRs (C layout: m = v + 8*hf, n = base+lrow).
  float hy[NTW][8], hz[NTW][8], ftr[NTW][8];
#pragma unroll
  for (int j = 0; j < NTW; ++j)
#pragma unroll
    for (int v = 0; v < 8; ++v) { hy[j][v] = 0.f; hz[j][v] = 0.f; ftr[j][v] = 0.f; }

  const float dt        = 0.01f;
  const float lif_keep  = 1.0f - 0.01f / 20.0f;     // 1 - dt/tau_m
  const float ft_decay  = 0.9990004998333750f;      // exp(-dt/10)
  const float ref_decay = 0.9607894391523232f;      // exp(-dt/0.25)

  __syncthreads();

#pragma unroll 1
  for (int t = 0; t < SEQL; ++t) {
    v8f acc[NTW] = {};

    // ---- GEMM with explicit depth-2 pipeline: WMMA(buf0) overlaps loads(buf1)
    ABFrag a0, a1, b0[NTW], b1[NTW];
    load_a(sbuf, lrow, hf, 0, a0);
    load_b(bpack, wave, lane, 0, b0);
#pragma unroll 1
    for (int kt = 0; kt < NKT; kt += 2) {
      load_a(sbuf, lrow, hf, kt + 1, a1);
      load_b(bpack, wave, lane, kt + 1, b1);
      do_wmma(a0, b0, acc);
      if (kt + 2 < NKT) {
        load_a(sbuf, lrow, hf, kt + 2, a0);
        load_b(bpack, wave, lane, kt + 2, b0);
      }
      do_wmma(a1, b1, acc);
    }
    __syncthreads();   // all spike reads done before update overwrites sbuf

    // Input scalars for this lane's 8 rows (16 unique addrs, L0/L2 broadcast).
    float xv[8];
#pragma unroll
    for (int v = 0; v < 8; ++v)
      xv[v] = x[(size_t)(rb + v + 8 * hf) * SEQL + t];

    // ---- fused elementwise state update; spikes written back into sbuf ----
#pragma unroll
    for (int j = 0; j < NTW; ++j) {
      int n = (wave * NTW + j) * 16 + lrow;
#pragma unroll
      for (int v = 0; v < 8; ++v) {
        int   m   = v + 8 * hf;
        F2    rsv = rs_l[m * PST + n];          // x=rf, y=fts
        float lvv = lv_l[m * LST + n];
        float I   = xv[v] * xw[j] + acc[j][v] + bv[j];
        lvv       = lvv * lif_keep + dt * I;
        float ls  = (lvv > 1.0f) ? 1.0f : 0.0f;
        lv_l[m * LST + n] = lvv - ls;
        float z   = hz[j][v] + dt * (sg * ls - gam[j] * hy[j][v] - ev[j] * hz[j][v]);
        float y   = hy[j][v] + dt * z;
        hz[j][v] = z; hy[j][v] = y;
        float s  = ((y - 1.0f - rsv.x) > 0.0f) ? 1.0f : 0.0f;
        float f  = ftr[j][v] * ft_decay + s;
        ftr[j][v] = f;
        rs_l[m * PST + n] = F2{rsv.x * ref_decay + s, rsv.y + f};
        sbuf[sw_addr(m, n)] = (s != 0.0f) ? (unsigned short)0x3F80
                                          : (unsigned short)0;
      }
    }
    __syncthreads();   // spikes for step t+1 visible to all waves
  }

  // out = concat(fts/L, ft) along columns
#pragma unroll
  for (int j = 0; j < NTW; ++j) {
    int n = (wave * NTW + j) * 16 + lrow;
#pragma unroll
    for (int v = 0; v < 8; ++v) {
      int    m = v + 8 * hf;
      size_t b = rb + m;
      out[b * (2 * NHID) + n]        = rs_l[m * PST + n].y * (1.0f / (float)SEQL);
      out[b * (2 * NHID) + NHID + n] = ftr[j][v];
    }
  }
}

extern "C" void kernel_launch(void* const* d_in, const int* in_sizes, int n_in,
                              void* d_out, int out_size, void* d_ws, size_t ws_size,
                              hipStream_t stream) {
  (void)in_sizes; (void)n_in; (void)out_size; (void)ws_size;
  const float* x    = (const float*)d_in[0];
  const float* x2h  = (const float*)d_in[1];
  const float* h2h  = (const float*)d_in[2];
  const float* bias = (const float*)d_in[3];
  const float* gam  = (const float*)d_in[4];
  const float* eps  = (const float*)d_in[5];
  const float* sg   = (const float*)d_in[6];
  unsigned short* bpack = (unsigned short*)d_ws;   // 2 MB bf16 fragment image of h2h

  coesn_prep_b<<<(NHID * NHID) / 256, 256, 0, stream>>>(h2h, bpack);
  coesn_persistent<<<BATCH / MT, NTHREADS, 0, stream>>>(
      x, x2h, bias, gam, eps, sg, bpack, (float*)d_out);
}
